// MultiHeadAttention_9371618640104
// MI455X (gfx1250) — matrix operand order repaired and checked by reading, not compiled, not run on hardware
//
#include <hip/hip_runtime.h>

// ---------------------------------------------------------------------------
// MultiHeadAttention for MI455X (gfx1250, wave32, WMMA + TDM).
//   D_MODEL=1024, N_HEAD=16, HEAD_DIM=64, B=4, S=2048  (fp32 in/out)
// Pipeline: fp32->f16 convert, 3 projection GEMMs (TDM-staged LDS, WMMA f16,
// f32 accum, head-split epilogues), fused flash-attention (TDM-staged K/V,
// WMMA + online softmax), output GEMM.
// ---------------------------------------------------------------------------

typedef __attribute__((ext_vector_type(16))) _Float16 v16h;
typedef __attribute__((ext_vector_type(8)))  float    v8f;
typedef __attribute__((ext_vector_type(4)))  float    f32x4;
typedef __attribute__((ext_vector_type(4)))  _Float16 h16x4;
typedef __attribute__((ext_vector_type(4)))  unsigned int u32x4;
typedef __attribute__((ext_vector_type(8)))  int      i32x8;
typedef __attribute__((ext_vector_type(4)))  int      i32x4;

#define D_MODEL  1024
#define N_HEAD   16
#define HEAD_DIM 64
#define BATCH    4
#define SEQ      2048
#define M_TOTAL  (BATCH * SEQ)

#if __has_builtin(__builtin_amdgcn_sched_barrier)
#define SCHED_FENCE() __builtin_amdgcn_sched_barrier(0)
#else
#define SCHED_FENCE()
#endif

union AFrag { v16h v; f32x4 q[2]; };

static __device__ __forceinline__ v8f wmma_f16(v16h a, v16h b, v8f c) {
  // 8 args: (neg_a, A, neg_b, B, c_mod, C, reuse_a, reuse_b)
  return __builtin_amdgcn_wmma_f32_16x16x32_f16(false, a, false, b,
                                                (short)0, c, false, false);
}

// ---------------------------------------------------------------------------
// TDM: issue a 2D f16 tile DMA  global -> LDS.
//   tile_w elements contiguous in memory (dim0), tile_h rows (dim1),
//   row stride in elements. One instruction per issuing wave (EXEC ignored),
//   tracked on TENSORcnt.
// Descriptor packing per CDNA5 ISA.
// ---------------------------------------------------------------------------
static __device__ __forceinline__ void tdm_load_2d_f16(
    unsigned int lds_byte_addr, const void* gptr, unsigned int tile_w,
    unsigned int tile_h, unsigned int row_stride_elems) {
  const unsigned long long ga = (unsigned long long)(uintptr_t)gptr;
  const unsigned int dim0 = 1u << 20;  // large: tiles never go OOB
  const unsigned int dim1 = 1u << 20;

  u32x4 g0;
  g0[0] = 1u;                                   // count=1 (valid), user mode
  g0[1] = lds_byte_addr;                        // lds_addr (bytes)
  g0[2] = (unsigned int)ga;                     // global_addr[31:0]
  g0[3] = (unsigned int)(ga >> 32) | (2u << 30);// global_addr[56:32], type=2

  i32x8 g1;
  g1[0] = (int)(1u << 16);                      // data_size=1 (2 bytes)
  g1[1] = (int)((dim0 & 0xffffu) << 16);        // tensor_dim0[15:0]
  g1[2] = (int)(((dim0 >> 16) & 0xffffu) | ((dim1 & 0xffffu) << 16));
  g1[3] = (int)(((dim1 >> 16) & 0xffffu) | ((tile_w & 0xffffu) << 16));
  g1[4] = (int)(tile_h & 0xffffu);              // tile_dim1 (tile_dim2=0)
  g1[5] = (int)row_stride_elems;                // tensor_dim0_stride[31:0]
  g1[6] = 0;                                    // stride0[47:32], stride1 lo
  g1[7] = 0;                                    // stride1 hi

  i32x4 z4 = {0, 0, 0, 0};
#if __has_include(<hip/amd_detail/amd_gfx1250_TDM.h>)
  i32x8 z8 = {0, 0, 0, 0, 0, 0, 0, 0};
  __builtin_amdgcn_tensor_load_to_lds(g0, g1, z4, z4, z8, 0);
#else
  __builtin_amdgcn_tensor_load_to_lds(g0, g1, z4, z4, 0);
#endif
}

static __device__ __forceinline__ unsigned int lds_addr_of(const void* p) {
  // LDS aperture occupies the upper 32 bits of the flat address (ISA §10.2);
  // the low 32 bits are the wave-relative LDS byte offset.
  return (unsigned int)(uintptr_t)p;
}

// ---------------------------------------------------------------------------
// fp32 -> f16 conversion, 4 elements per thread
// ---------------------------------------------------------------------------
__global__ __launch_bounds__(256) void cvt_f32_to_f16(
    const float* __restrict__ src, _Float16* __restrict__ dst, int n4) {
  int i = blockIdx.x * blockDim.x + threadIdx.x;
  if (i < n4) {
    f32x4 v = ((const f32x4*)src)[i];
    ((h16x4*)dst)[i] = __builtin_convertvector(v, h16x4);
  }
}

// ---------------------------------------------------------------------------
// NT GEMM (TDM-staged):  Y[M][N] = A[M][K] * Bw[N][K]^T + bias[N]
// Block = 8 waves -> 64(M) x 128(N), wave tile 16x64 (4x v8f accumulators).
// Per 64-wide K-chunk, wave 0 DMAs A(64x64) and B(128x64) f16 tiles into a
// double-buffered LDS stage via tensor_load_to_lds.  All chunk fragments are
// preloaded into VGPRs (pinned by a sched_barrier), then the 8 WMMAs issue
// back-to-back behind a single ds wait.  DMA of chunk i+1 overlaps compute
// of chunk i.
// mode 0: fp32 [M][N];  mode 1: f16 Qh/Kh [b][h][s][hd];
// mode 2: f16 Vt [b][h][hd][s]
// ---------------------------------------------------------------------------
__global__ __launch_bounds__(256) void gemm_nt(
    const _Float16* __restrict__ A, const _Float16* __restrict__ Bw,
    const float* __restrict__ bias, void* __restrict__ outp,
    int M, int N, int K, int mode) {
  __shared__ _Float16 Ablk[2][64][64];    // 16 KB
  __shared__ _Float16 Bblk[2][128][64];   // 32 KB

  const int lane = threadIdx.x & 31;
  const int wave = threadIdx.x >> 5;   // 0..7
  const int col  = lane & 15;
  const int g    = lane >> 4;
  const int mw   = wave >> 1;          // 0..3 (M sub-tile)
  const int nw   = wave & 1;           // 0..1 (N sub-tile)

  const int mB = blockIdx.x * 64;      // block M origin
  const int nB = blockIdx.y * 128;     // block N origin

  v8f zero = {};
  v8f acc[4];
#pragma unroll
  for (int t = 0; t < 4; ++t) acc[t] = zero;

  const int nChunks = K / 64;

  // Prologue: DMA chunk 0 into buffer 0.
  if (wave == 0) {
    tdm_load_2d_f16(lds_addr_of(&Ablk[0][0][0]), A + (size_t)mB * K, 64, 64, K);
    tdm_load_2d_f16(lds_addr_of(&Bblk[0][0][0]), Bw + (size_t)nB * K, 64, 128, K);
  }

  for (int c = 0; c < nChunks; ++c) {
    const int buf = c & 1;
    if (wave == 0) __builtin_amdgcn_s_wait_tensorcnt(0);  // chunk c resident
    __syncthreads();  // publish chunk c; all reads of buffer buf^1 done
    if (wave == 0 && (c + 1) < nChunks) {
      const int k1 = (c + 1) * 64;
      tdm_load_2d_f16(lds_addr_of(&Ablk[buf ^ 1][0][0]),
                      A + (size_t)mB * K + k1, 64, 64, K);
      tdm_load_2d_f16(lds_addr_of(&Bblk[buf ^ 1][0][0]),
                      Bw + (size_t)nB * K + k1, 64, 128, K);
    }

    // Preload the whole chunk's fragments, then 8 back-to-back WMMAs.
    const _Float16* arow = &Ablk[buf][mw * 16 + col][0];
    AFrag a[2];
    a[0].q[0] = *(const f32x4*)(arow + 16 * g);
    a[0].q[1] = *(const f32x4*)(arow + 16 * g + 8);
    a[1].q[0] = *(const f32x4*)(arow + 32 + 16 * g);
    a[1].q[1] = *(const f32x4*)(arow + 32 + 16 * g + 8);
    AFrag b[8];
#pragma unroll
    for (int t = 0; t < 4; ++t) {
      const _Float16* brow = &Bblk[buf][nw * 64 + 16 * t + col][0];
      b[2 * t + 0].q[0] = *(const f32x4*)(brow + 16 * g);
      b[2 * t + 0].q[1] = *(const f32x4*)(brow + 16 * g + 8);
      b[2 * t + 1].q[0] = *(const f32x4*)(brow + 32 + 16 * g);
      b[2 * t + 1].q[1] = *(const f32x4*)(brow + 32 + 16 * g + 8);
    }
    SCHED_FENCE();  // keep all ds_loads above, all WMMAs below (one ds wait)
#pragma unroll
    for (int t = 0; t < 4; ++t) {
      acc[t] = wmma_f16(a[0].v, b[2 * t + 0].v, acc[t]);
      acc[t] = wmma_f16(a[1].v, b[2 * t + 1].v, acc[t]);
    }
    SCHED_FENCE();
  }

  const int m0 = mB + mw * 16;
  const int n0 = nB + nw * 64;
#pragma unroll
  for (int t = 0; t < 4; ++t) {
    const int n  = n0 + 16 * t + col;
    const float bv = bias[n];
#pragma unroll
    for (int r = 0; r < 8; ++r) {
      const int m   = m0 + r + 8 * g;   // C layout: M = r + 8*(lane/16)
      const float v = acc[t][r] + bv;
      if (mode == 0) {
        ((float*)outp)[(size_t)m * N + n] = v;
      } else {
        const int b2 = m / SEQ, s = m % SEQ;
        const int h = n / HEAD_DIM, d = n % HEAD_DIM;
        if (mode == 1)
          ((_Float16*)outp)[((size_t)(b2 * N_HEAD + h) * SEQ + s) * HEAD_DIM + d] =
              (_Float16)v;
        else
          ((_Float16*)outp)[((size_t)(b2 * N_HEAD + h) * HEAD_DIM + d) * SEQ + s] =
              (_Float16)v;
      }
    }
  }
}

// ---------------------------------------------------------------------------
// Fused flash attention per (b,h).  Block = 4 waves, each wave owns 16 q-rows
// (64 q-rows per block).  K/V 64x64 tiles are TDM-staged into double-buffered
// LDS (issued by wave 0, shared by all 4 waves); online softmax; P tile
// bounced through per-wave LDS to convert C-layout -> A-layout.
// ---------------------------------------------------------------------------
__global__ __launch_bounds__(128) void flash_attn(
    const _Float16* __restrict__ Qh,   // [B*H][S][64]
    const _Float16* __restrict__ Kh,   // [B*H][S][64]
    const _Float16* __restrict__ Vt,   // [B*H][64][S]
    _Float16* __restrict__ Ctx) {      // [B][S][D_MODEL]
  __shared__ _Float16 Kblk[2][64][64];  // 16 KB
  __shared__ _Float16 Vblk[2][64][64];  // 16 KB
  __shared__ _Float16 pbuf[4][16][64];  // 8 KB

  const int lane = threadIdx.x & 31;
  const int wave = threadIdx.x >> 5;   // 0..3
  const int col  = lane & 15;
  const int g    = lane >> 4;

  const int bh = blockIdx.x;           // 0..63
  const int b  = bh >> 4;
  const int h  = bh & 15;
  const int q0 = blockIdx.y * 64 + wave * 16;

  const _Float16* Qp = Qh + (size_t)bh * SEQ * HEAD_DIM;
  const _Float16* Kp = Kh + (size_t)bh * SEQ * HEAD_DIM;
  const _Float16* Vp = Vt + (size_t)bh * HEAD_DIM * SEQ;

  // Q A-fragments for both 32-wide K-steps of the hd=64 contraction.
  const _Float16* qrow = Qp + (size_t)(q0 + col) * HEAD_DIM;
  AFrag qa0, qa1;
  qa0.q[0] = *(const f32x4*)(qrow + 16 * g);
  qa0.q[1] = *(const f32x4*)(qrow + 16 * g + 8);
  qa1.q[0] = *(const f32x4*)(qrow + 32 + 16 * g);
  qa1.q[1] = *(const f32x4*)(qrow + 32 + 16 * g + 8);

  v8f zero = {};
  v8f ctx[4];
#pragma unroll
  for (int t = 0; t < 4; ++t) ctx[t] = zero;
  float m_i[8], l_i[8];
#pragma unroll
  for (int r = 0; r < 8; ++r) { m_i[r] = -1e30f; l_i[r] = 0.0f; }

  const float scale = 0.125f;  // 1/sqrt(64)
  const int nTiles = SEQ / 64;

  // Prologue: DMA K/V tile 0 into buffer 0.
  if (wave == 0) {
    tdm_load_2d_f16(lds_addr_of(&Kblk[0][0][0]), Kp, 64, 64, HEAD_DIM);
    tdm_load_2d_f16(lds_addr_of(&Vblk[0][0][0]), Vp, 64, 64, SEQ);
  }

  for (int c = 0; c < nTiles; ++c) {
    const int buf = c & 1;
    if (wave == 0) __builtin_amdgcn_s_wait_tensorcnt(0);  // tile c resident
    __syncthreads();  // publish tile c; all reads of buffer buf^1 done
    if (wave == 0 && (c + 1) < nTiles) {
      const int kb1 = (c + 1) * 64;
      tdm_load_2d_f16(lds_addr_of(&Kblk[buf ^ 1][0][0]),
                      Kp + (size_t)kb1 * HEAD_DIM, 64, 64, HEAD_DIM);
      tdm_load_2d_f16(lds_addr_of(&Vblk[buf ^ 1][0][0]),
                      Vp + kb1, 64, 64, SEQ);
    }

    // ---- scores: 16(q) x 64(k) -- preload 8 B-fragments, then 8 WMMAs ----
    AFrag bk[8];
#pragma unroll
    for (int t = 0; t < 4; ++t) {
      const _Float16* krow = &Kblk[buf][16 * t + col][0];
      bk[2 * t + 0].q[0] = *(const f32x4*)(krow + 16 * g);
      bk[2 * t + 0].q[1] = *(const f32x4*)(krow + 16 * g + 8);
      bk[2 * t + 1].q[0] = *(const f32x4*)(krow + 32 + 16 * g);
      bk[2 * t + 1].q[1] = *(const f32x4*)(krow + 32 + 16 * g + 8);
    }
    SCHED_FENCE();  // batch the ds_loads; WMMAs issue behind one wait
    v8f cs[4];
#pragma unroll
    for (int t = 0; t < 4; ++t) cs[t] = zero;
#pragma unroll
    for (int t = 0; t < 4; ++t) {
      cs[t] = wmma_f16(qa0.v, bk[2 * t + 0].v, cs[t]);
      cs[t] = wmma_f16(qa1.v, bk[2 * t + 1].v, cs[t]);
    }
    SCHED_FENCE();

    // ---- online softmax (rows live across 16-lane half-groups) ----
#pragma unroll
    for (int t = 0; t < 4; ++t)
#pragma unroll
      for (int r = 0; r < 8; ++r) cs[t][r] *= scale;

#pragma unroll
    for (int r = 0; r < 8; ++r) {
      float m = fmaxf(fmaxf(cs[0][r], cs[1][r]), fmaxf(cs[2][r], cs[3][r]));
      m = fmaxf(m, __shfl_xor(m, 1));
      m = fmaxf(m, __shfl_xor(m, 2));
      m = fmaxf(m, __shfl_xor(m, 4));
      m = fmaxf(m, __shfl_xor(m, 8));
      const float mnew  = fmaxf(m_i[r], m);
      const float alpha = __expf(m_i[r] - mnew);
      m_i[r] = mnew;
      float rs = 0.0f;
#pragma unroll
      for (int t = 0; t < 4; ++t) {
        const float p = __expf(cs[t][r] - mnew);
        cs[t][r] = p;
        rs += p;
      }
      rs += __shfl_xor(rs, 1);
      rs += __shfl_xor(rs, 2);
      rs += __shfl_xor(rs, 4);
      rs += __shfl_xor(rs, 8);
      l_i[r] = l_i[r] * alpha + rs;
#pragma unroll
      for (int t = 0; t < 4; ++t) ctx[t][r] *= alpha;
    }

    // ---- P tile: C-layout regs -> LDS (row-major 16x64 f16) ----
#pragma unroll
    for (int t = 0; t < 4; ++t)
#pragma unroll
      for (int r = 0; r < 8; ++r)
        pbuf[wave][r + 8 * g][16 * t + col] = (_Float16)cs[t][r];

    // ---- reload P in A-fragment layout; preload V B-fragments ----
    AFrag pa0, pa1;
    pa0.q[0] = *(const f32x4*)&pbuf[wave][col][16 * g];
    pa0.q[1] = *(const f32x4*)&pbuf[wave][col][16 * g + 8];
    pa1.q[0] = *(const f32x4*)&pbuf[wave][col][32 + 16 * g];
    pa1.q[1] = *(const f32x4*)&pbuf[wave][col][32 + 16 * g + 8];

    AFrag bvf[8];
#pragma unroll
    for (int t = 0; t < 4; ++t) {
      const _Float16* vrow = &Vblk[buf][16 * t + col][0];
      bvf[2 * t + 0].q[0] = *(const f32x4*)(vrow + 16 * g);
      bvf[2 * t + 0].q[1] = *(const f32x4*)(vrow + 16 * g + 8);
      bvf[2 * t + 1].q[0] = *(const f32x4*)(vrow + 32 + 16 * g);
      bvf[2 * t + 1].q[1] = *(const f32x4*)(vrow + 32 + 16 * g + 8);
    }
    SCHED_FENCE();  // batch P + V ds_loads before the 8 context WMMAs
#pragma unroll
    for (int t = 0; t < 4; ++t) {
      ctx[t] = wmma_f16(pa0.v, bvf[2 * t + 0].v, ctx[t]);
      ctx[t] = wmma_f16(pa1.v, bvf[2 * t + 1].v, ctx[t]);
    }
    SCHED_FENCE();
  }

  // ---- epilogue: normalize and store merged-head layout ----
#pragma unroll
  for (int t = 0; t < 4; ++t) {
    const int d = 16 * t + col;
#pragma unroll
    for (int r = 0; r < 8; ++r) {
      const int s   = q0 + r + 8 * g;
      const float v = ctx[t][r] / l_i[r];
      Ctx[(size_t)(b * SEQ + s) * D_MODEL + h * HEAD_DIM + d] = (_Float16)v;
    }
  }
}

// ---------------------------------------------------------------------------
// Host-side launcher
// ---------------------------------------------------------------------------
extern "C" void kernel_launch(void* const* d_in, const int* in_sizes, int n_in,
                              void* d_out, int out_size, void* d_ws,
                              size_t ws_size, hipStream_t stream) {
  const float* q  = (const float*)d_in[0];
  const float* k  = (const float*)d_in[1];
  const float* v  = (const float*)d_in[2];
  const float* Wq = (const float*)d_in[3];
  const float* bq = (const float*)d_in[4];
  const float* Wk = (const float*)d_in[5];
  const float* bk = (const float*)d_in[6];
  const float* Wv = (const float*)d_in[7];
  const float* bv = (const float*)d_in[8];
  const float* Wo = (const float*)d_in[9];
  const float* bo = (const float*)d_in[10];
  float* out = (float*)d_out;

  // Workspace layout (f16 elements); total 60M halves = 120 MB.
  const size_t MB = 1024 * 1024;  // 1M elements
  _Float16* wsh  = (_Float16*)d_ws;
  _Float16* Wq16 = wsh + 0 * MB;
  _Float16* Wk16 = wsh + 1 * MB;
  _Float16* Wv16 = wsh + 2 * MB;
  _Float16* Wo16 = wsh + 3 * MB;
  _Float16* qf   = wsh + 4 * MB;    // [8192][1024]
  _Float16* kf   = wsh + 12 * MB;
  _Float16* vf   = wsh + 20 * MB;
  _Float16* Qh   = wsh + 28 * MB;   // [B*H][S][64]
  _Float16* Kh   = wsh + 36 * MB;
  _Float16* Vt   = wsh + 44 * MB;   // [B*H][64][S]
  _Float16* ctx  = wsh + 52 * MB;   // [8192][1024]

  const int TPB = 256;
  auto cvt = [&](const float* s, _Float16* d, int n) {
    const int n4 = n / 4;
    cvt_f32_to_f16<<<(n4 + TPB - 1) / TPB, TPB, 0, stream>>>(s, d, n4);
  };
  cvt(Wq, Wq16, D_MODEL * D_MODEL);
  cvt(Wk, Wk16, D_MODEL * D_MODEL);
  cvt(Wv, Wv16, D_MODEL * D_MODEL);
  cvt(Wo, Wo16, D_MODEL * D_MODEL);
  cvt(q, qf, M_TOTAL * D_MODEL);
  cvt(k, kf, M_TOTAL * D_MODEL);
  cvt(v, vf, M_TOTAL * D_MODEL);

  dim3 ggrid(M_TOTAL / 64, D_MODEL / 128);
  gemm_nt<<<ggrid, 256, 0, stream>>>(qf, Wq16, bq, Qh, M_TOTAL, D_MODEL,
                                     D_MODEL, 1);
  gemm_nt<<<ggrid, 256, 0, stream>>>(kf, Wk16, bk, Kh, M_TOTAL, D_MODEL,
                                     D_MODEL, 1);
  gemm_nt<<<ggrid, 256, 0, stream>>>(vf, Wv16, bv, Vt, M_TOTAL, D_MODEL,
                                     D_MODEL, 2);

  dim3 agrid(BATCH * N_HEAD, SEQ / 64);
  flash_attn<<<agrid, 128, 0, stream>>>(Qh, Kh, Vt, ctx);

  gemm_nt<<<ggrid, 256, 0, stream>>>(ctx, Wo16, bo, out, M_TOTAL, D_MODEL,
                                     D_MODEL, 0);
}
